// Compress_30047591202836
// MI455X (gfx1250) — compile-verified, matrix operand order checked by reading, not run on hardware
//
#include <hip/hip_runtime.h>
#include <hip/hip_bf16.h>

// ---------------------------------------------------------------------------
// Problem constants (match reference)
// ---------------------------------------------------------------------------
constexpr int S_   = 2048;
constexpr int H_   = 2048;
constexpr int NH_  = 32;
constexpr int HD_  = 64;
constexpr int KVD_ = 512;     // NKV * HD
constexpr int COMP_= 1024;
constexpr int FF_  = 5632;
constexpr int TOT_ = 3072;    // S + COMP
constexpr int DEPTH_ = 2;

typedef __bf16          bf16x16 __attribute__((ext_vector_type(16)));
typedef float           f32x8   __attribute__((ext_vector_type(8)));
typedef unsigned int    u32x4   __attribute__((ext_vector_type(4)));
typedef unsigned int    u32x2   __attribute__((ext_vector_type(2)));
typedef int             i32x2   __attribute__((ext_vector_type(2)));
typedef unsigned short  u16x16  __attribute__((ext_vector_type(16)));

#if __has_builtin(__builtin_amdgcn_global_load_async_to_lds_b64)
#define GEMM_ASYNC 1
#else
#define GEMM_ASYNC 0
#endif

union Frag {
    bf16x16 bf;
    u32x4   q[2];   // two 16-byte chunks (8 bf16 each)
    u16x16  h;
};

__device__ __forceinline__ unsigned short f2bfu(float f) {
    union { float f; unsigned u; } x; x.f = f;
    unsigned u = x.u;
    unsigned r = (u + 0x7FFFu + ((u >> 16) & 1u)) >> 16;   // round-nearest-even
    return (unsigned short)r;
}

template <int N>
__device__ __forceinline__ void wait_async() {
#if __has_builtin(__builtin_amdgcn_s_wait_asynccnt)
    __builtin_amdgcn_s_wait_asynccnt(N);
#else
    asm volatile("s_wait_asynccnt %0" ::"i"(N));
#endif
}

// ---------------------------------------------------------------------------
// fp32 -> bf16 convert (grid-stride)
// ---------------------------------------------------------------------------
__global__ void k_cvt_bf16(const float* __restrict__ in,
                           unsigned short* __restrict__ out, long n) {
    long i = (long)blockIdx.x * blockDim.x + threadIdx.x;
    long st = (long)gridDim.x * blockDim.x;
    for (; i < n; i += st) out[i] = f2bfu(in[i]);
}

// ---------------------------------------------------------------------------
// transpose + convert: in (rows x cols) f32 -> out (cols x rows) bf16
// ---------------------------------------------------------------------------
__global__ void k_transpose_cvt(const float* __restrict__ in,
                                unsigned short* __restrict__ out,
                                int rows, int cols) {
    __shared__ float tile[32][33];
    int bx = blockIdx.x * 32;   // col base
    int by = blockIdx.y * 32;   // row base
    for (int i = threadIdx.y; i < 32; i += 8)
        tile[i][threadIdx.x] = in[(size_t)(by + i) * cols + bx + threadIdx.x];
    __syncthreads();
    for (int i = threadIdx.y; i < 32; i += 8)
        out[(size_t)(bx + i) * rows + by + threadIdx.x] = f2bfu(tile[threadIdx.x][i]);
}

// ---------------------------------------------------------------------------
// RMSNorm over rows of (rows x cols): writes f32 and bf16 outputs
// ---------------------------------------------------------------------------
__global__ void k_rmsnorm(const float* __restrict__ x, const float* __restrict__ w,
                          float* __restrict__ oF, unsigned short* __restrict__ oB,
                          int cols) {
    __shared__ float red[256];
    int row = blockIdx.x;
    const float* xr = x + (size_t)row * cols;
    float ss = 0.f;
    for (int i = threadIdx.x; i < cols; i += 256) { float v = xr[i]; ss += v * v; }
    red[threadIdx.x] = ss;
    __syncthreads();
    for (int st = 128; st > 0; st >>= 1) {
        if (threadIdx.x < st) red[threadIdx.x] += red[threadIdx.x + st];
        __syncthreads();
    }
    float inv = rsqrtf(red[0] / (float)cols + 1e-6f);
    for (int i = threadIdx.x; i < cols; i += 256) {
        float v = xr[i] * inv * w[i];
        oF[(size_t)row * cols + i] = v;
        oB[(size_t)row * cols + i] = f2bfu(v);
    }
}

// ---------------------------------------------------------------------------
// silu(g) * u  ->  bf16
// ---------------------------------------------------------------------------
__global__ void k_silu_mul(const float* __restrict__ g, const float* __restrict__ u,
                           unsigned short* __restrict__ out, long n) {
    long i = (long)blockIdx.x * blockDim.x + threadIdx.x;
    long st = (long)gridDim.x * blockDim.x;
    for (; i < n; i += st) {
        float gv = g[i];
        float s  = gv / (1.f + __expf(-gv));
        out[i] = f2bfu(s * u[i]);
    }
}

// ---------------------------------------------------------------------------
// WMMA GEMM v2:  C[M,N] = A[M,K](bf16 rm) . W[N,K](bf16 rm)^T
//   Block = 256 threads (8 waves), tile = 32(M) x 256(N).
//   Wave (mw = w&1, nw = w>>1) computes 16(M) x 64(N) -> 4 f32x8 accumulators:
//   one A fragment (from LDS, double-buffered, async-staged) feeds 4 WMMAs.
//   A tile staged memory->LDS via global_load_async_to_lds_b64 when available
//   (ASYNCcnt + s_wait_asynccnt), else plain load+ds_store fallback.
//   LDS row pitch = 40 ushorts (80B) => conflict-free ds_load_b128 fragments.
// ---------------------------------------------------------------------------
__global__ __launch_bounds__(256)
void k_gemm_wmma(const unsigned short* __restrict__ A,
                 const unsigned short* __restrict__ W,
                 float* __restrict__ outF,
                 unsigned short* __restrict__ outB,
                 const float* __restrict__ resid,
                 const float* __restrict__ biasM,
                 int M, int N, int K) {
    __shared__ unsigned short As[2][32][40];

    const int tid  = threadIdx.x;
    const int lane = tid & 31;
    const int wv   = tid >> 5;
    const int r    = lane & 15;
    const int sel  = lane >> 4;
    const int mw   = wv & 1;          // 0..1  : M sub-tile
    const int nw   = wv >> 1;         // 0..3  : N group of 64
    const int m0   = blockIdx.y * 32;
    const int n0   = blockIdx.x * 256;
    const int ncol0= n0 + nw * 64;

    // staging: thread copies 8 bytes of the 32x32 A tile per k-step
    const int srow = tid >> 3;          // 0..31
    const int sseg = (tid & 7) * 4;     // ushort col {0,4,...,28}
    const unsigned short* astage = A + (size_t)(m0 + srow) * K + sseg;

    const unsigned short* brow = W + (size_t)(ncol0 + r) * K;
    f32x8 acc[4] = {};

    const int nk = K / 32;
    auto stage = [&](int buf, int k0) {
#if GEMM_ASYNC
        __builtin_amdgcn_global_load_async_to_lds_b64(
            (i32x2*)(astage + k0),
            (i32x2*)&As[buf][srow][sseg], 0, 0);
#else
        *(u32x2*)&As[buf][srow][sseg] = *(const u32x2*)(astage + k0);
#endif
    };

    stage(0, 0);
    for (int it = 0; it < nk; ++it) {
        const int cur = it & 1;
        const int k0  = it * 32;
        if (it + 1 < nk) stage(cur ^ 1, k0 + 32);
#if GEMM_ASYNC
        if (it + 1 < nk) wait_async<1>(); else wait_async<0>();
#endif
        __syncthreads();

        // A fragment from LDS (ISA 16-bit A striping)
        Frag a;
        const u32x4* ap = (const u32x4*)&As[cur][mw * 16 + r][sel * 8];
        a.q[0] = ap[0]; a.q[1] = ap[2];

        // pull the weight stream into GL2 ahead of use
        __builtin_prefetch(brow + k0 + 8 * 32, 0, 0);

#pragma unroll
        for (int t = 0; t < 4; ++t) {
            Frag b;
            const u32x4* bp = (const u32x4*)(brow + (size_t)t * 16 * K + k0 + sel * 8);
            b.q[0] = bp[0]; b.q[1] = bp[2];
            acc[t] = __builtin_amdgcn_wmma_f32_16x16x32_bf16(false, a.bf, false, b.bf,
                                                             (short)0, acc[t], false, false);
        }
        __syncthreads();
    }

#pragma unroll
    for (int t = 0; t < 4; ++t) {
#pragma unroll
        for (int g = 0; g < 8; ++g) {
            int m = m0 + mw * 16 + g + sel * 8;
            int n = ncol0 + t * 16 + r;
            float v = acc[t][g];
            if (biasM) v += biasM[m];
            if (resid) v += resid[(size_t)m * N + n];
            if (outF)  outF[(size_t)m * N + n] = v;
            if (outB)  outB[(size_t)m * N + n] = f2bfu(v);
        }
    }
}

// ---------------------------------------------------------------------------
// Fused GQA flash attention.
//   Q: (COMP x H) bf16, heads along H (head*64..).  K,V: (TOT x KVD) bf16.
//   O: (COMP x H) bf16.  scale = 1/sqrt(64) = 0.125.  head -> kv head = head/4.
// grid = (NH, COMP/16), block = 32 (one wave per 16-row q tile).
// ---------------------------------------------------------------------------
__global__ void k_flash_attn(const unsigned short* __restrict__ Q,
                             const unsigned short* __restrict__ Kb,
                             const unsigned short* __restrict__ Vb,
                             unsigned short* __restrict__ O) {
    __shared__ unsigned short pbuf[16 * 32];
    const int lane = threadIdx.x & 31;
    const int r    = lane & 15;
    const int sel  = lane >> 4;
    const int head = blockIdx.x;
    const int q0   = blockIdx.y * 16;
    const int kvh  = head >> 2;            // GROUPS = 4 (repeat_interleave)

    // Q tile 16x64 as two A-fragments (K steps 0 and 32)
    Frag qa[2];
    {
        const unsigned short* qp = Q + (size_t)(q0 + r) * H_ + head * HD_;
#pragma unroll
        for (int ks = 0; ks < 2; ++ks) {
            const u32x4* p = (const u32x4*)(qp + ks * 32 + sel * 8);
            qa[ks].q[0] = p[0]; qa[ks].q[1] = p[2];
        }
    }

    float Mrow[8], Lrow[8];
    f32x8 oacc[4];
#pragma unroll
    for (int g = 0; g < 8; ++g) { Mrow[g] = -1e30f; Lrow[g] = 0.f; }
#pragma unroll
    for (int t = 0; t < 4; ++t) oacc[t] = (f32x8){};

    for (int j0 = 0; j0 < TOT_; j0 += 32) {
        // ---- scores: two 16x16 tiles over the 32-kv chunk -----------------
        f32x8 s[2];
#pragma unroll
        for (int t = 0; t < 2; ++t) {
            f32x8 c = {};
#pragma unroll
            for (int ks = 0; ks < 2; ++ks) {
                Frag kb;
                const u32x4* p = (const u32x4*)(Kb + (size_t)(j0 + t * 16 + r) * KVD_
                                                + kvh * HD_ + ks * 32 + sel * 8);
                kb.q[0] = p[0]; kb.q[1] = p[2];
                c = __builtin_amdgcn_wmma_f32_16x16x32_bf16(false, qa[ks].bf, false, kb.bf,
                                                            (short)0, c, false, false);
            }
#pragma unroll
            for (int g = 0; g < 8; ++g) s[t][g] = c[g] * 0.125f;
        }

        // ---- online softmax row stats (half-wave reductions) --------------
        float mnew[8], alpha[8];
#pragma unroll
        for (int g = 0; g < 8; ++g) {
            float mx = fmaxf(s[0][g], s[1][g]);
            for (int off = 1; off < 16; off <<= 1)
                mx = fmaxf(mx, __shfl_xor(mx, off, 32));
            mnew[g]  = fmaxf(Mrow[g], mx);
            alpha[g] = __expf(Mrow[g] - mnew[g]);
            Mrow[g]  = mnew[g];
        }

        // ---- P = exp(s - m), rowsum, stash P (bf16) to LDS ----------------
#pragma unroll
        for (int g = 0; g < 8; ++g) {
            float p0 = __expf(s[0][g] - mnew[g]);
            float p1 = __expf(s[1][g] - mnew[g]);
            float rs = p0 + p1;
            for (int off = 1; off < 16; off <<= 1)
                rs += __shfl_xor(rs, off, 32);
            Lrow[g] = Lrow[g] * alpha[g] + rs;
            int m = g + sel * 8;
            pbuf[m * 32 + r]      = f2bfu(p0);
            pbuf[m * 32 + 16 + r] = f2bfu(p1);
        }
        asm volatile("s_wait_dscnt 0" ::: "memory");

        // ---- reload P with A-fragment striping ----------------------------
        Frag pa;
        {
            const u32x4* pv = (const u32x4*)(pbuf + r * 32 + sel * 8);
            pa.q[0] = pv[0]; pa.q[1] = pv[2];
        }
        asm volatile("s_wait_dscnt 0" ::: "memory");

        // ---- rescale O, accumulate P(16x32) . V(32x64) --------------------
#pragma unroll
        for (int t = 0; t < 4; ++t) {
#pragma unroll
            for (int g = 0; g < 8; ++g) oacc[t][g] *= alpha[g];
            Frag vb;
#pragma unroll
            for (int e = 0; e < 16; ++e) {
                int kk = (e < 8) ? (sel * 8 + e) : (16 + sel * 8 + (e - 8));
                vb.h[e] = Vb[(size_t)(j0 + kk) * KVD_ + kvh * HD_ + t * 16 + r];
            }
            oacc[t] = __builtin_amdgcn_wmma_f32_16x16x32_bf16(false, pa.bf, false, vb.bf,
                                                              (short)0, oacc[t], false, false);
        }
    }

    // ---- epilogue: O / L, store bf16 --------------------------------------
#pragma unroll
    for (int t = 0; t < 4; ++t) {
#pragma unroll
        for (int g = 0; g < 8; ++g) {
            float v = oacc[t][g] / Lrow[g];
            O[(size_t)(q0 + g + sel * 8) * H_ + head * HD_ + t * 16 + r] = f2bfu(v);
        }
    }
}

// ---------------------------------------------------------------------------
// Host orchestration
// ---------------------------------------------------------------------------
extern "C" void kernel_launch(void* const* d_in, const int* in_sizes, int n_in,
                              void* d_out, int out_size, void* d_ws, size_t ws_size,
                              hipStream_t stream) {
    const float* hidden   = (const float*)d_in[0];
    const float* comp_w   = (const float*)d_in[1];
    const float* comp_b   = (const float*)d_in[2];
    const float* q_w      = (const float*)d_in[3];
    const float* k_w      = (const float*)d_in[4];
    const float* v_w      = (const float*)d_in[5];
    const float* o_w      = (const float*)d_in[6];
    const float* attn_nw  = (const float*)d_in[7];
    const float* mlp_nw   = (const float*)d_in[8];
    const float* gate_w   = (const float*)d_in[9];
    const float* up_w     = (const float*)d_in[10];
    const float* down_w   = (const float*)d_in[11];

    char* p = (char*)d_ws;
    auto alloc = [&](size_t bytes) -> char* {
        char* r = p; p += (bytes + 255) & ~(size_t)255; return r;
    };
    auto* hiddenT_bf = (unsigned short*)alloc((size_t)H_ * S_ * 2);
    auto* residual_bf= (unsigned short*)alloc((size_t)TOT_ * H_ * 2);
    auto* comp_w_bf  = (unsigned short*)alloc((size_t)COMP_ * S_ * 2);
    auto* q_w_bf     = (unsigned short*)alloc((size_t)H_ * H_ * 2);
    auto* k_w_bf     = (unsigned short*)alloc((size_t)KVD_ * H_ * 2);
    auto* v_w_bf     = (unsigned short*)alloc((size_t)KVD_ * H_ * 2);
    auto* o_w_bf     = (unsigned short*)alloc((size_t)H_ * H_ * 2);
    auto* gate_w_bf  = (unsigned short*)alloc((size_t)FF_ * H_ * 2);
    auto* up_w_bf    = (unsigned short*)alloc((size_t)FF_ * H_ * 2);
    auto* down_w_bf  = (unsigned short*)alloc((size_t)H_ * FF_ * 2);
    auto* compressedF= (float*)alloc((size_t)COMP_ * H_ * 4);
    auto* ctF        = (float*)alloc((size_t)COMP_ * H_ * 4);
    auto* ctB        = (unsigned short*)alloc((size_t)COMP_ * H_ * 2);
    auto* q_bf       = (unsigned short*)alloc((size_t)COMP_ * H_ * 2);
    auto* k_bf       = (unsigned short*)alloc((size_t)TOT_ * KVD_ * 2);
    auto* v_bf       = (unsigned short*)alloc((size_t)TOT_ * KVD_ * 2);
    auto* attn_bf    = (unsigned short*)alloc((size_t)COMP_ * H_ * 2);
    auto* gateF      = (float*)alloc((size_t)COMP_ * FF_ * 4);
    auto* upF        = (float*)alloc((size_t)COMP_ * FF_ * 4);
    auto* gu_bf      = (unsigned short*)alloc((size_t)COMP_ * FF_ * 2);
    (void)ws_size; (void)n_in; (void)in_sizes; (void)out_size;

    auto cvt = [&](const float* in, unsigned short* out, long n) {
        k_cvt_bf16<<<4096, 256, 0, stream>>>(in, out, n);
    };
    auto gemm = [&](const unsigned short* A, const unsigned short* W,
                    float* oF, unsigned short* oB,
                    const float* resid, const float* bias,
                    int M, int N, int K) {
        dim3 g(N / 256, M / 32);
        k_gemm_wmma<<<g, 256, 0, stream>>>(A, W, oF, oB, resid, bias, M, N, K);
    };

    // ---- precompute bf16 operands -----------------------------------------
    cvt(hidden, residual_bf, (long)S_ * H_);                 // rows 0..S-1 of residual
    k_transpose_cvt<<<dim3(H_ / 32, S_ / 32), dim3(32, 8), 0, stream>>>(
        hidden, hiddenT_bf, S_, H_);
    cvt(comp_w, comp_w_bf, (long)COMP_ * S_);
    cvt(q_w,    q_w_bf,    (long)H_ * H_);
    cvt(k_w,    k_w_bf,    (long)KVD_ * H_);
    cvt(v_w,    v_w_bf,    (long)KVD_ * H_);
    cvt(o_w,    o_w_bf,    (long)H_ * H_);
    cvt(gate_w, gate_w_bf, (long)FF_ * H_);
    cvt(up_w,   up_w_bf,   (long)FF_ * H_);
    cvt(down_w, down_w_bf, (long)H_ * FF_);

    // compressed[c,h] = sum_s comp_w[c,s] * hiddenT[h,s]  (+ comp_b[c])
    gemm(comp_w_bf, hiddenT_bf, compressedF, nullptr, nullptr, comp_b,
         COMP_, H_, S_);

    for (int d = 0; d < DEPTH_; ++d) {
        // residual rows S..S+COMP-1 = bf16(compressed)
        cvt(compressedF, residual_bf + (size_t)S_ * H_, (long)COMP_ * H_);

        // ct = rmsnorm(compressed, attn_norm_w)
        k_rmsnorm<<<COMP_, 256, 0, stream>>>(compressedF, attn_nw, ctF, ctB, H_);

        // q/k/v projections
        gemm(ctB,         q_w_bf, nullptr, q_bf, nullptr, nullptr, COMP_, H_,   H_);
        gemm(residual_bf, k_w_bf, nullptr, k_bf, nullptr, nullptr, TOT_,  KVD_, H_);
        gemm(residual_bf, v_w_bf, nullptr, v_bf, nullptr, nullptr, TOT_,  KVD_, H_);

        // fused GQA flash attention
        k_flash_attn<<<dim3(NH_, COMP_ / 16), 32, 0, stream>>>(q_bf, k_bf, v_bf, attn_bf);

        // compressed = attn @ o_w^T + ct
        gemm(attn_bf, o_w_bf, compressedF, nullptr, ctF, nullptr, COMP_, H_, H_);

        // ct2 = rmsnorm(compressed, mlp_norm_w)   (reuse ct buffers)
        k_rmsnorm<<<COMP_, 256, 0, stream>>>(compressedF, mlp_nw, ctF, ctB, H_);

        // gated MLP
        gemm(ctB, gate_w_bf, gateF, nullptr, nullptr, nullptr, COMP_, FF_, H_);
        gemm(ctB, up_w_bf,   upF,   nullptr, nullptr, nullptr, COMP_, FF_, H_);
        k_silu_mul<<<4096, 256, 0, stream>>>(gateF, upF, gu_bf, (long)COMP_ * FF_);

        float* mlpOut = (d == DEPTH_ - 1) ? (float*)d_out : compressedF;
        gemm(gu_bf, down_w_bf, mlpOut, nullptr, ctF, nullptr, COMP_, H_, FF_);
    }
}
